// AxialAttentionBlock_17789754540112
// MI455X (gfx1250) — matrix operand order repaired and checked by reading, not compiled, not run on hardware
//
#include <hip/hip_runtime.h>
#include <math.h>

// ---------------------------------------------------------------------------
// Types for CDNA5 WMMA (wave32): D(16x16,f32) = A(16x32,bf16) x B(32x16,bf16) + C
// ---------------------------------------------------------------------------
typedef __attribute__((ext_vector_type(16))) __bf16        bf16x16;
typedef __attribute__((ext_vector_type(8)))  float         f32x8;
typedef __attribute__((ext_vector_type(4)))  unsigned int  u32x4;

union Frag16 { bf16x16 v; u32x4 q[2]; };

__device__ __forceinline__ unsigned short f2bf(float f) {
  unsigned int u = __float_as_uint(f);
  u += 0x7FFFu + ((u >> 16) & 1u);   // round-to-nearest-even
  return (unsigned short)(u >> 16);
}
__device__ __forceinline__ float bf2f(unsigned short h) {
  return __uint_as_float(((unsigned int)h) << 16);
}
__device__ __forceinline__ f32x8 wmma_bf16(bf16x16 a, bf16x16 b, f32x8 c) {
  // (neg_a, A, neg_b, B, c_mod, C, reuse_a, reuse_b)
  return __builtin_amdgcn_wmma_f32_16x16x32_bf16(false, a, false, b, (short)0, c,
                                                 false, false);
}
__device__ __forceinline__ f32x8 zero8() {
  f32x8 z;
#pragma unroll
  for (int i = 0; i < 8; ++i) z[i] = 0.0f;
  return z;
}

// ---------------------------------------------------------------------------
// Problem constants
// ---------------------------------------------------------------------------
#define BB    2
#define LL    96          // H == W == 96
#define CC    512
#define NH    16          // heads
#define HD    32          // head dim
#define SP    (LL*LL)     // 9216 spatial positions per batch
#define MTOT  (BB*SP)     // 18432 rows

// ---------------------------------------------------------------------------
// Small elementwise / reduction kernels
// ---------------------------------------------------------------------------
__global__ void zero_f32(float* p, int n) {
  int i = blockIdx.x * blockDim.x + threadIdx.x;
  if (i < n) p[i] = 0.0f;
}

// per-(b,c) sum / sumsq over spatial dims; grid (B, C/256, 16), block 256
__global__ void stats_accum(const float* __restrict__ src,
                            float* __restrict__ sum, float* __restrict__ sumsq) {
  int c = blockIdx.y * 256 + threadIdx.x;
  int b = blockIdx.x;
  int z = blockIdx.z;
  float s = 0.f, ss = 0.f;
  for (int i = 0; i < SP / 16; ++i) {
    int sp = z * (SP / 16) + i;
    float v = src[((size_t)b * SP + sp) * CC + c];
    s += v; ss += v * v;
  }
  atomicAdd(&sum[b * CC + c], s);
  atomicAdd(&sumsq[b * CC + c], ss);
}

// rstd = 1/(std+eps) with ddof=1 (RMSInstanceNorm2d), eps = 1e-8
__global__ void stats_final(const float* __restrict__ sum,
                            const float* __restrict__ sumsq,
                            float* __restrict__ rstd) {
  int i = blockIdx.x * blockDim.x + threadIdx.x;
  if (i >= BB * CC) return;
  const float n = (float)SP;
  float mu  = sum[i] / n;
  float var = (sumsq[i] - sum[i] * mu) / (n - 1.0f);
  var = fmaxf(var, 0.0f);
  rstd[i] = 1.0f / (sqrtf(var) + 1e-8f);
}

__global__ void cast_bf16(const float* __restrict__ in,
                          unsigned short* __restrict__ out, int n) {
  int i = blockIdx.x * blockDim.x + threadIdx.x;
  if (i < n) out[i] = f2bf(in[i]);
}

// y = x * rstd[b,c] * w[c]  -> bf16
__global__ void norm_cast(const float* __restrict__ x,
                          const float* __restrict__ rstd,
                          const float* __restrict__ w,
                          unsigned short* __restrict__ out) {
  size_t i = (size_t)blockIdx.x * 256 + threadIdx.x;
  int c = (int)(i & (CC - 1));
  int b = (int)(i / ((size_t)SP * CC));
  out[i] = f2bf(x[i] * rstd[b * CC + c] * w[c]);
}

// out = x1 + m * rstd[b,c] * mlpn_w[c] * gamma_mlp[c]
__global__ void final_add(const float* __restrict__ x1, const float* __restrict__ m,
                          const float* __restrict__ rstd,
                          const float* __restrict__ mlpw,
                          const float* __restrict__ gmlp,
                          float* __restrict__ out) {
  size_t i = (size_t)blockIdx.x * 256 + threadIdx.x;
  int c = (int)(i & (CC - 1));
  int b = (int)(i / ((size_t)SP * CC));
  out[i] = x1[i] + m[i] * rstd[b * CC + c] * mlpw[c] * gmlp[c];
}

// ---------------------------------------------------------------------------
// T5 bucketed relative-position bias tables: bias[head][q][k]
// ---------------------------------------------------------------------------
__device__ __forceinline__ int pymod(int a, int b) {
  int r = a % b;
  if (r != 0 && ((r < 0) != (b < 0))) r += b;
  return r;
}

__global__ void build_bias(const float* __restrict__ emb, const int* __restrict__ bcs,
                           float* __restrict__ biasx, float* __restrict__ biasy) {
  int t = blockIdx.x * blockDim.x + threadIdx.x;
  const int total = 2 * NH * LL * LL;
  if (t >= total) return;
  int k     = t % LL;
  int q     = (t / LL) % LL;
  int head  = (t / (LL * LL)) % NH;
  int which = t / (NH * LL * LL);
  int bc    = bcs[which];           // bcs[0,0] / bcs[0,1]
  int rel = k - q;
  if (bc == 1) {
    int thresh = LL / 2;
    int rw = rel;
    if (rel < -thresh) rw = pymod(rel, thresh);
    if (rw > thresh)   rw = pymod(rw, -thresh);
    rel = rw;
  }
  int n = -rel;
  const int nb = 16;                // NUM_BUCKETS / 2
  int ret = (n < 0) ? nb : 0;
  n = (n < 0) ? -n : n;
  int bucket;
  if (n < 8) {
    bucket = ret + n;
  } else {
    float lf = logf((float)(n < 1 ? 1 : n) / 8.0f) / logf(4.0f) * 8.0f;
    int vl = 8 + (int)lf;
    if (vl > 15) vl = 15;
    bucket = ret + vl;
  }
  float v = emb[bucket * NH + head];
  float* dst = (which == 0) ? biasx : biasy;
  dst[((size_t)head * LL + q) * LL + k] = v;
}

// ---------------------------------------------------------------------------
// QKV post-process: per-(b,s,head) LayerNorm of q,k (eps 1e-5), fold 1/sqrt(hd)
// into q; scatter v into transposed layouts for row / column attention.
// ---------------------------------------------------------------------------
__global__ void qkv_post(const unsigned short* __restrict__ qkv,
                         const float* __restrict__ qnw, const float* __restrict__ qnb,
                         const float* __restrict__ knw, const float* __restrict__ knb,
                         unsigned short* __restrict__ Qb, unsigned short* __restrict__ Kb,
                         unsigned short* __restrict__ Vtr, unsigned short* __restrict__ Vtc) {
  int t = blockIdx.x * blockDim.x + threadIdx.x;
  if (t >= BB * SP * NH) return;
  int head = t % NH;
  int s    = (t / NH) % SP;
  int b    = t / (SP * NH);
  const unsigned short* src = qkv + ((size_t)b * SP + s) * (3 * CC) + head * (3 * HD);

  float qv[HD], kv[HD];
  float qs = 0.f, qss = 0.f, ks = 0.f, kss = 0.f;
#pragma unroll
  for (int d = 0; d < HD; ++d) {
    float q = bf2f(src[d]);
    float k = bf2f(src[HD + d]);
    qv[d] = q; kv[d] = k;
    qs += q; qss += q * q; ks += k; kss += k * k;
  }
  float qmu = qs / (float)HD, kmu = ks / (float)HD;
  float qvar = qss / (float)HD - qmu * qmu;
  float kvar = kss / (float)HD - kmu * kmu;
  float qr = rsqrtf(fmaxf(qvar, 0.f) + 1e-5f);
  float kr = rsqrtf(fmaxf(kvar, 0.f) + 1e-5f);
  const float scale = 0.17677669529663687f;  // 32^-0.5

  unsigned short* qd = Qb + ((size_t)b * SP + s) * CC + head * HD;
  unsigned short* kd = Kb + ((size_t)b * SP + s) * CC + head * HD;
  int h = s / LL, wc = s % LL;
  size_t vr = (((size_t)(b * LL + h)  * NH + head) * HD) * LL + wc;  // Vtr[b][h][head][d][w]
  size_t vc = (((size_t)(b * LL + wc) * NH + head) * HD) * LL + h;   // Vtc[b][w][head][d][h]
#pragma unroll
  for (int d = 0; d < HD; ++d) {
    qd[d] = f2bf(((qv[d] - qmu) * qr * qnw[d] + qnb[d]) * scale);
    kd[d] = f2bf((kv[d] - kmu) * kr * knw[d] + knb[d]);
    unsigned short vvv = src[2 * HD + d];
    Vtr[vr + (size_t)d * LL] = vvv;
    Vtc[vc + (size_t)d * LL] = vvv;
  }
}

// ---------------------------------------------------------------------------
// Generic WMMA GEMM:  C(MxN) = A(MxK, bf16 rowmajor) * W(NxK, bf16 rowmajor)^T + bias
// epi: 0 = bias only; 1 = bias + exact GELU; 2 = (bias) * gamma[n] + res[m][n]
// Block 128x128, 8 waves in 4(M)x2(N), wave tile 32x64 (2x4 WMMA tiles).
// ---------------------------------------------------------------------------
__global__ __launch_bounds__(256) void gemm_bf16_nt(
    const unsigned short* __restrict__ A, int lda,
    const unsigned short* __restrict__ W, int ldb,
    const float* __restrict__ bias,
    float* __restrict__ Cf, unsigned short* __restrict__ Cb, int ldc,
    int K, int epi,
    const float* __restrict__ gamma, const float* __restrict__ res) {
  const int lane = threadIdx.x & 31;
  const int wid  = threadIdx.x >> 5;
  const int wm = wid >> 1;
  const int wn = wid & 1;
  const int mbase = blockIdx.y * 128 + wm * 32;
  const int nbase = blockIdx.x * 128 + wn * 64;
  const int l15 = lane & 15;
  const int hiA = (lane >= 16) ? 8 : 0;    // A frag: K chunks {hiA..hiA+7, hiA+16..hiA+23}
  const int hiB = (lane >= 16) ? 8 : 0;    // B frag: K chunks {hiB..hiB+7, hiB+16..hiB+23}

  f32x8 acc[2][4];
#pragma unroll
  for (int mt = 0; mt < 2; ++mt)
#pragma unroll
    for (int nt = 0; nt < 4; ++nt) acc[mt][nt] = zero8();

  const unsigned short* arow[2];
#pragma unroll
  for (int mt = 0; mt < 2; ++mt)
    arow[mt] = A + (size_t)(mbase + mt * 16 + l15) * lda + hiA;
  const unsigned short* brow[4];
#pragma unroll
  for (int nt = 0; nt < 4; ++nt)
    brow[nt] = W + (size_t)(nbase + nt * 16 + l15) * ldb + hiB;

  for (int k0 = 0; k0 < K; k0 += 32) {
    Frag16 af[2];
#pragma unroll
    for (int mt = 0; mt < 2; ++mt) {
      af[mt].q[0] = *(const u32x4*)(arow[mt] + k0);
      af[mt].q[1] = *(const u32x4*)(arow[mt] + k0 + 16);
    }
    Frag16 bfr[4];
#pragma unroll
    for (int nt = 0; nt < 4; ++nt) {
      bfr[nt].q[0] = *(const u32x4*)(brow[nt] + k0);
      bfr[nt].q[1] = *(const u32x4*)(brow[nt] + k0 + 16);
    }
#pragma unroll
    for (int mt = 0; mt < 2; ++mt)
#pragma unroll
      for (int nt = 0; nt < 4; ++nt)
        acc[mt][nt] = wmma_bf16(af[mt].v, bfr[nt].v, acc[mt][nt]);
  }

  const int roff = (lane >= 16) ? 8 : 0;
#pragma unroll
  for (int mt = 0; mt < 2; ++mt)
#pragma unroll
    for (int nt = 0; nt < 4; ++nt) {
      const int n = nbase + nt * 16 + l15;
      const float bs = bias ? bias[n] : 0.0f;
#pragma unroll
      for (int r = 0; r < 8; ++r) {
        const int m = mbase + mt * 16 + r + roff;
        float v = acc[mt][nt][r] + bs;
        if (epi == 1) {
          v = 0.5f * v * (1.0f + erff(v * 0.70710678118654752f));
        } else if (epi == 2) {
          v = v * gamma[n] + res[(size_t)m * ldc + n];
        }
        const size_t idx = (size_t)m * ldc + n;
        if (Cf) Cf[idx] = v;
        if (Cb) Cb[idx] = f2bf(v);
      }
    }
}

// ---------------------------------------------------------------------------
// Axial attention: one workgroup (6 waves) per (b, line, head).
// Wave w owns queries [16w, 16w+16): S = Q K^T (+bias), softmax, O = P V.
// mode 0: row attention (queries along W), writes 0.5*O
// mode 1: column attention (queries along H), accumulates += 0.5*O
// ---------------------------------------------------------------------------
__global__ __launch_bounds__(192) void attn_axial(
    const unsigned short* __restrict__ Qb, const unsigned short* __restrict__ Kb,
    const unsigned short* __restrict__ Vt, const float* __restrict__ biasT,
    float* __restrict__ Aout, int mode) {
  const int lane = threadIdx.x & 31;
  const int w    = threadIdx.x >> 5;        // m-tile 0..5
  const int p    = blockIdx.x;
  const int head = p & 15;
  const int line = (p >> 4) % LL;
  const int b    = p / (NH * LL);

  size_t qbase; int qstride; size_t out_off; int out_qstride;
  if (mode == 0) {  // row: fixed h=line, queries/keys over w
    qbase       = ((size_t)(b * LL + line) * LL) * CC + head * HD;
    qstride     = CC;
    out_off     = (size_t)b * SP * CC + (size_t)line * LL * CC + head * HD;
    out_qstride = CC;
  } else {          // col: fixed w=line, queries/keys over h
    qbase       = ((size_t)(b * LL) * LL + line) * CC + head * HD;
    qstride     = LL * CC;
    out_off     = (size_t)b * SP * CC + (size_t)line * CC + head * HD;
    out_qstride = LL * CC;
  }
  const size_t vtbase = (((size_t)(b * LL + line) * NH + head) * HD) * LL;

  __shared__ __align__(16) unsigned short P[6][16][LL];

  const int l15 = lane & 15;
  const int hiA = (lane >= 16) ? 8 : 0;
  const int hiB = (lane >= 16) ? 8 : 0;
  const int roff = (lane >= 16) ? 8 : 0;

  // Q fragment for this wave's 16 rows (K dim = HD = 32, one fragment)
  Frag16 qf;
  {
    const unsigned short* ptr = Qb + qbase + (size_t)(w * 16 + l15) * qstride + hiA;
    qf.q[0] = *(const u32x4*)ptr;
    qf.q[1] = *(const u32x4*)(ptr + 16);
  }

  // S = Q K^T : 6 tiles of 16x16, one WMMA each
  f32x8 sacc[6];
#pragma unroll
  for (int nt = 0; nt < 6; ++nt) {
    Frag16 kf;
    const unsigned short* ptr = Kb + qbase + (size_t)(nt * 16 + l15) * qstride + hiB;
    kf.q[0] = *(const u32x4*)ptr;
    kf.q[1] = *(const u32x4*)(ptr + 16);
    sacc[nt] = wmma_bf16(qf.v, kf.v, zero8());
  }

  // bias + softmax per query row; rows 0-7 in lanes 0-15, rows 8-15 in lanes 16-31
  const int mrow_off = w * 16 + roff;
#pragma unroll
  for (int r = 0; r < 8; ++r) {
    float sv[6];
    float mx = -3.0e38f;
#pragma unroll
    for (int nt = 0; nt < 6; ++nt) {
      float s = sacc[nt][r] +
                biasT[((size_t)head * LL + (mrow_off + r)) * LL + nt * 16 + l15];
      sv[nt] = s;
      mx = fmaxf(mx, s);
    }
#pragma unroll
    for (int m2 = 1; m2 < 16; m2 <<= 1) mx = fmaxf(mx, __shfl_xor(mx, m2, 32));
    float sum = 0.0f;
#pragma unroll
    for (int nt = 0; nt < 6; ++nt) { float e = __expf(sv[nt] - mx); sv[nt] = e; sum += e; }
#pragma unroll
    for (int m2 = 1; m2 < 16; m2 <<= 1) sum += __shfl_xor(sum, m2, 32);
    float inv = 1.0f / sum;
#pragma unroll
    for (int nt = 0; nt < 6; ++nt)
      P[w][r + roff][nt * 16 + l15] = f2bf(sv[nt] * inv);
  }

  __syncthreads();

  // O = P V : 2 n-tiles (HD=32), K dim = 96 -> 3 WMMAs each
#pragma unroll
  for (int nt2 = 0; nt2 < 2; ++nt2) {
    f32x8 acc = zero8();
#pragma unroll
    for (int ks = 0; ks < 3; ++ks) {
      Frag16 fa;
      const unsigned short* ap = &P[w][l15][ks * 32 + hiA];
      fa.q[0] = *(const u32x4*)ap;
      fa.q[1] = *(const u32x4*)(ap + 16);
      Frag16 fb;
      const unsigned short* bp = Vt + vtbase + (size_t)(nt2 * 16 + l15) * LL + ks * 32 + hiB;
      fb.q[0] = *(const u32x4*)bp;
      fb.q[1] = *(const u32x4*)(bp + 16);
      acc = wmma_bf16(fa.v, fb.v, acc);
    }
#pragma unroll
    for (int r = 0; r < 8; ++r) {
      const int m = w * 16 + r + roff;            // query index within line
      const int d = nt2 * 16 + l15;               // head-dim index
      const size_t idx = out_off + (size_t)m * out_qstride + d;
      const float v = acc[r] * 0.5f;
      if (mode == 0) Aout[idx] = v; else Aout[idx] += v;
    }
  }
}

// ---------------------------------------------------------------------------
// Host-side launch sequence
// ---------------------------------------------------------------------------
extern "C" void kernel_launch(void* const* d_in, const int* in_sizes, int n_in,
                              void* d_out, int out_size, void* d_ws, size_t ws_size,
                              hipStream_t stream) {
  (void)in_sizes; (void)n_in; (void)out_size; (void)ws_size;

  const float* x       = (const float*)d_in[0];
  const float* norm1_w = (const float*)d_in[1];
  const float* in_w    = (const float*)d_in[2];
  const float* in_b    = (const float*)d_in[3];
  const float* qn_w    = (const float*)d_in[4];
  const float* qn_b    = (const float*)d_in[5];
  const float* kn_w    = (const float*)d_in[6];
  const float* kn_b    = (const float*)d_in[7];
  const float* emb     = (const float*)d_in[8];
  const float* norm2_w = (const float*)d_in[9];
  const float* out_w   = (const float*)d_in[10];
  const float* out_b   = (const float*)d_in[11];
  const float* gatt    = (const float*)d_in[12];
  const float* fc1_w   = (const float*)d_in[13];
  const float* fc1_b   = (const float*)d_in[14];
  const float* fc2_w   = (const float*)d_in[15];
  const float* fc2_b   = (const float*)d_in[16];
  const float* mlpn_w  = (const float*)d_in[17];
  const float* gmlp    = (const float*)d_in[18];
  const int*   bcs     = (const int*)d_in[19];

  char* base = (char*)d_ws;
  size_t off = 0;
  auto take = [&](size_t bytes) -> void* {
    void* r = base + off;
    off = (off + bytes + 255) & ~(size_t)255;
    return r;
  };
  float* sum   = (float*)take((size_t)BB * CC * 4);
  float* sumsq = (float*)take((size_t)BB * CC * 4);
  float* rstd1 = (float*)take((size_t)BB * CC * 4);
  float* rstd2 = (float*)take((size_t)BB * CC * 4);
  float* rstdm = (float*)take((size_t)BB * CC * 4);
  float* biasx = (float*)take((size_t)NH * LL * LL * 4);
  float* biasy = (float*)take((size_t)NH * LL * LL * 4);
  unsigned short* w_in  = (unsigned short*)take((size_t)3 * CC * CC * 2);
  unsigned short* w_out = (unsigned short*)take((size_t)CC * CC * 2);
  unsigned short* w_f1  = (unsigned short*)take((size_t)4 * CC * CC * 2);
  unsigned short* w_f2  = (unsigned short*)take((size_t)4 * CC * CC * 2);
  unsigned short* xn    = (unsigned short*)take((size_t)MTOT * CC * 2);
  unsigned short* qkvb  = (unsigned short*)take((size_t)MTOT * 3 * CC * 2);
  unsigned short* Qb    = (unsigned short*)take((size_t)MTOT * CC * 2);
  unsigned short* Kb    = (unsigned short*)take((size_t)MTOT * CC * 2);
  unsigned short* Vtr   = (unsigned short*)take((size_t)MTOT * CC * 2);
  unsigned short* Vtc   = (unsigned short*)take((size_t)MTOT * CC * 2);
  float*          abuf  = (float*)take((size_t)MTOT * CC * 4);
  unsigned short* Anb   = (unsigned short*)take((size_t)MTOT * CC * 2);
  float*          x1f   = (float*)take((size_t)MTOT * CC * 4);
  unsigned short* x1b   = (unsigned short*)take((size_t)MTOT * CC * 2);
  unsigned short* hmid  = (unsigned short*)take((size_t)MTOT * 4 * CC * 2);
  float*          mbuf  = (float*)take((size_t)MTOT * CC * 4);

  const dim3 blk256(256);
  const dim3 gElem((unsigned)((size_t)MTOT * CC / 256));  // 36864 blocks

  // ---- norm1 stats over x ----
  zero_f32<<<dim3(4), blk256, 0, stream>>>(sum, BB * CC);
  zero_f32<<<dim3(4), blk256, 0, stream>>>(sumsq, BB * CC);
  stats_accum<<<dim3(BB, CC / 256, 16), blk256, 0, stream>>>(x, sum, sumsq);
  stats_final<<<dim3(4), blk256, 0, stream>>>(sum, sumsq, rstd1);

  // ---- weight casts + bias tables ----
  cast_bf16<<<dim3((3 * CC * CC + 255) / 256), blk256, 0, stream>>>(in_w, w_in, 3 * CC * CC);
  cast_bf16<<<dim3((CC * CC + 255) / 256), blk256, 0, stream>>>(out_w, w_out, CC * CC);
  cast_bf16<<<dim3((4 * CC * CC + 255) / 256), blk256, 0, stream>>>(fc1_w, w_f1, 4 * CC * CC);
  cast_bf16<<<dim3((4 * CC * CC + 255) / 256), blk256, 0, stream>>>(fc2_w, w_f2, 4 * CC * CC);
  build_bias<<<dim3((2 * NH * LL * LL + 255) / 256), blk256, 0, stream>>>(emb, bcs, biasx, biasy);

  // ---- xn = RMSInstNorm(x) -> bf16 ----
  norm_cast<<<gElem, blk256, 0, stream>>>(x, rstd1, norm1_w, xn);

  // ---- QKV projection GEMM (18432 x 512 x 1536) ----
  gemm_bf16_nt<<<dim3(12, 144), blk256, 0, stream>>>(
      xn, CC, w_in, CC, in_b, nullptr, qkvb, 3 * CC, CC, 0, nullptr, nullptr);

  // ---- per-head LayerNorm of q,k (+scale); V transposes ----
  qkv_post<<<dim3((BB * SP * NH + 255) / 256), blk256, 0, stream>>>(
      qkvb, qn_w, qn_b, kn_w, kn_b, Qb, Kb, Vtr, Vtc);

  // ---- axial attention: rows then columns ----
  attn_axial<<<dim3(BB * LL * NH), dim3(192), 0, stream>>>(Qb, Kb, Vtr, biasx, abuf, 0);
  attn_axial<<<dim3(BB * LL * NH), dim3(192), 0, stream>>>(Qb, Kb, Vtc, biasy, abuf, 1);

  // ---- norm2 + out projection fused with gamma_att + residual ----
  zero_f32<<<dim3(4), blk256, 0, stream>>>(sum, BB * CC);
  zero_f32<<<dim3(4), blk256, 0, stream>>>(sumsq, BB * CC);
  stats_accum<<<dim3(BB, CC / 256, 16), blk256, 0, stream>>>(abuf, sum, sumsq);
  stats_final<<<dim3(4), blk256, 0, stream>>>(sum, sumsq, rstd2);
  norm_cast<<<gElem, blk256, 0, stream>>>(abuf, rstd2, norm2_w, Anb);
  gemm_bf16_nt<<<dim3(4, 144), blk256, 0, stream>>>(
      Anb, CC, w_out, CC, out_b, x1f, x1b, CC, CC, 2, gatt, x);

  // ---- MLP: fc1 (+exact GELU) then fc2 ----
  gemm_bf16_nt<<<dim3(16, 144), blk256, 0, stream>>>(
      x1b, CC, w_f1, CC, fc1_b, nullptr, hmid, 4 * CC, CC, 1, nullptr, nullptr);
  gemm_bf16_nt<<<dim3(4, 144), blk256, 0, stream>>>(
      hmid, 4 * CC, w_f2, 4 * CC, fc2_b, mbuf, nullptr, CC, 4 * CC, 0, nullptr, nullptr);

  // ---- mlp norm + final residual ----
  zero_f32<<<dim3(4), blk256, 0, stream>>>(sum, BB * CC);
  zero_f32<<<dim3(4), blk256, 0, stream>>>(sumsq, BB * CC);
  stats_accum<<<dim3(BB, CC / 256, 16), blk256, 0, stream>>>(mbuf, sum, sumsq);
  stats_final<<<dim3(4), blk256, 0, stream>>>(sum, sumsq, rstdm);
  final_add<<<gElem, blk256, 0, stream>>>(x1f, mbuf, rstdm, mlpn_w, gmlp, (float*)d_out);
}
